// PWQuad_8581344657568
// MI455X (gfx1250) — matrix operand order repaired and checked by reading, not compiled, not run on hardware
//
#include <hip/hip_runtime.h>

// ---------------- types ----------------
typedef __attribute__((ext_vector_type(16))) __bf16        v16bf;
typedef __attribute__((ext_vector_type(8)))  float         v8f;
typedef __attribute__((ext_vector_type(4)))  unsigned int  u32x4;

#define HIDN   256
#define OUTC   520
#define OUTP3  576          // w3 padded to 36 tiles of 16 (4-way n-split, even iters)
#define MT     128          // samples per workgroup
#define H1S    264          // bf16 row stride for h1/h2 in LDS (256 + 8 pad)
#define Z3S    529          // f32 row stride for z3 in LDS (odd -> conflict-free)
#define BN_EPS 1e-5f

// f32 -> bf16, round-to-nearest-even
__device__ __forceinline__ __bf16 f2bf(float x) {
    union { float f; unsigned int u; } a; a.f = x;
    unsigned int r = (a.u + 0x7FFFu + ((a.u >> 16) & 1u)) >> 16;
    union { unsigned short s; __bf16 b; } o; o.s = (unsigned short)r;
    return o.b;
}

union FragAB { v16bf v; u32x4 q[2]; };

// A fragment 16x32 bf16 from LDS tile (row-major, stride in bf16 elems).
// lane L: row = L&15, K-half = (L>>4)*8; elems 0..7 -> K kh+0..7, 8..15 -> K kh+16..23
__device__ __forceinline__ v16bf load_frag_A(const __bf16* base, int stride, int lane, int k0) {
    int r  = lane & 15;
    int kh = (lane >> 4) << 3;
    const __bf16* p = base + r * stride + k0 + kh;
    FragAB f;
    f.q[0] = *(const u32x4*)(p);
    f.q[1] = *(const u32x4*)(p + 16);
    return f.v;
}

// B fragment 32x16 bf16 from an LDS-staged strip laid out [16][256] row-major.
// lane L: col n = L&15, K-half = (L>>4)*8; elems 0..7 -> K kh+0..7, 8..15 -> K kh+16..23
__device__ __forceinline__ v16bf load_frag_Bs(const __bf16* strip, int lane, int k0) {
    int n  = lane & 15;
    int ks = k0 + ((lane >> 4) << 3);
    const __bf16* p = strip + n * HIDN + ks;
    FragAB f;
    f.q[0] = *(const u32x4*)(p);
    f.q[1] = *(const u32x4*)(p + 16);
    return f.v;
}

// generic pointer into LDS -> 32-bit wave-relative LDS byte offset (low 32 bits)
__device__ __forceinline__ unsigned ldsoff(const void* p) {
    return (unsigned)(unsigned long long)p;
}

// CDNA5 async global->LDS copy (ASYNCcnt), 16B per lane
__device__ __forceinline__ void async_ld_b128(unsigned lds_off, const void* gbase, unsigned voff) {
    asm volatile("global_load_async_to_lds_b128 %0, %1, %2 offset:0"
                 :: "v"(lds_off), "v"(voff), "s"(gbase) : "memory");
}
__device__ __forceinline__ void wait_async() {
    asm volatile("s_wait_asynccnt 0x0" ::: "memory");
}

__device__ __forceinline__ v8f bf16_wmma(v16bf a, v16bf b, v8f c) {
    return __builtin_amdgcn_wmma_f32_16x16x32_bf16(false, a, false, b, (short)0, c, false, false);
}

// ---------------- kernel 1: weight transpose / bf16 conversion ----------------
__global__ __launch_bounds__(256) void pwq_prepw(
    const float* __restrict__ w2, const float* __restrict__ w3, const float* __restrict__ b3,
    __bf16* __restrict__ w2T, __bf16* __restrict__ w3T, float* __restrict__ b3p)
{
    const int total = HIDN * HIDN + OUTP3 * HIDN + OUTP3;
    for (int i = blockIdx.x * blockDim.x + threadIdx.x; i < total; i += gridDim.x * blockDim.x) {
        if (i < HIDN * HIDN) {
            int n = i >> 8, k = i & 255;
            w2T[i] = f2bf(w2[k * HIDN + n]);
        } else if (i < HIDN * HIDN + OUTP3 * HIDN) {
            int o = i - HIDN * HIDN;
            int n = o >> 8, k = o & 255;
            w3T[o] = f2bf(n < OUTC ? w3[k * OUTC + n] : 0.f);
        } else {
            int n = i - HIDN * HIDN - OUTP3 * HIDN;
            b3p[n] = (n < OUTC) ? b3[n] : 0.f;
        }
    }
}

// ---------------- kernel 2: first/second moments of xA ----------------
__global__ __launch_bounds__(256) void pwq_stats0(
    const float* __restrict__ x, int B, float* __restrict__ S0, float* __restrict__ X0)
{
    __shared__ float sh[72];
    int tid = threadIdx.x;
    if (tid < 72) sh[tid] = 0.f;
    __syncthreads();
    float s[8]; float c[64];
#pragma unroll
    for (int p = 0; p < 8; p++) s[p] = 0.f;
#pragma unroll
    for (int p = 0; p < 64; p++) c[p] = 0.f;
    for (int i = blockIdx.x * blockDim.x + tid; i < B; i += gridDim.x * blockDim.x) {
        float v[8];
#pragma unroll
        for (int p = 0; p < 8; p++) { v[p] = x[(size_t)i * 17 + p]; s[p] += v[p]; }
#pragma unroll
        for (int p = 0; p < 8; p++)
#pragma unroll
            for (int q = 0; q < 8; q++) c[p * 8 + q] += v[p] * v[q];
    }
#pragma unroll
    for (int p = 0; p < 8; p++) atomicAdd(&sh[p], s[p]);
#pragma unroll
    for (int p = 0; p < 64; p++) atomicAdd(&sh[8 + p], c[p]);
    __syncthreads();
    if (tid < 8)       atomicAdd(&S0[tid], sh[tid]);
    else if (tid < 72) atomicAdd(&X0[tid - 8], sh[tid]);
}

// ---------------- kernel 3: fold BN0 + BN1 into effective layer-1 affine ----------------
__global__ __launch_bounds__(256) void pwq_prep01(
    const float* __restrict__ S0, const float* __restrict__ X0,
    const float* __restrict__ g0, const float* __restrict__ b0,
    const float* __restrict__ w1, const float* __restrict__ g1, const float* __restrict__ b1,
    float* __restrict__ W1e, float* __restrict__ C1e, float Binv)
{
    __shared__ float mean0[8], a0[8], c0[8], b0s[8], covH[64];
    int tid = threadIdx.x;
    if (tid < 8) mean0[tid] = S0[tid] * Binv;
    __syncthreads();
    if (tid < 8) {
        float var = X0[tid * 8 + tid] * Binv - mean0[tid] * mean0[tid];
        float a = g0[tid] * rsqrtf(var + BN_EPS);
        a0[tid] = a; c0[tid] = b0[tid] - a * mean0[tid]; b0s[tid] = b0[tid];
    }
    __syncthreads();
    if (tid < 64) {
        int p = tid >> 3, q = tid & 7;
        covH[tid] = (X0[tid] * Binv - mean0[p] * mean0[q]) * a0[p] * a0[q];
    }
    __syncthreads();
    float wj[8];
#pragma unroll
    for (int p = 0; p < 8; p++) wj[p] = w1[p * HIDN + tid];
    float mz = 0.f;
#pragma unroll
    for (int p = 0; p < 8; p++) mz += b0s[p] * wj[p];   // E[h0] == bn0_b exactly
    float vz = 0.f;
#pragma unroll
    for (int p = 0; p < 8; p++)
#pragma unroll
        for (int q = 0; q < 8; q++) vz += wj[p] * wj[q] * covH[p * 8 + q];
    float a1 = g1[tid] * rsqrtf(vz + BN_EPS);
    float c1 = b1[tid] - a1 * mz;
    float csum = 0.f;
#pragma unroll
    for (int p = 0; p < 8; p++) { W1e[p * HIDN + tid] = a0[p] * wj[p] * a1; csum += c0[p] * wj[p]; }
    C1e[tid] = a1 * csum + c1;
}

// ---------------- kernel 4: z2 column stats via staged WMMA (for BN2) ----------------
// dyn LDS: xt[128*18] f32 | h1[128*264] bf16 | bufB 32KB | cs[256] | cq[256]
__global__ __launch_bounds__(256) void pwq_stats2(
    const float* __restrict__ x, const float* __restrict__ W1e, const float* __restrict__ C1e,
    const __bf16* __restrict__ w2T, float* __restrict__ gsum, float* __restrict__ gsq)
{
    extern __shared__ char smem[];
    float*  xt   = (float*)smem;                         // 9216 B
    __bf16* h1   = (__bf16*)(smem + 9216);               // 67584 B
    char*   bufB = smem + 9216 + 67584;                  // 32768 B
    float*  cs   = (float*)(smem + 9216 + 67584 + 32768);
    float*  cq   = cs + HIDN;

    const int tid = threadIdx.x, lane = tid & 31, wv = tid >> 5;
    const int s0 = blockIdx.x * MT;
    cs[tid] = 0.f; cq[tid] = 0.f;
    for (int i = tid; i < MT * 17; i += 256) {
        int s = i / 17, c = i % 17;
        xt[s * 18 + c] = x[(size_t)(s0 + s) * 17 + c];
    }
    __syncthreads();
    { // layer 1 (K=8): one output column per thread
        float wcol[8];
#pragma unroll
        for (int p = 0; p < 8; p++) wcol[p] = W1e[p * HIDN + tid];
        float cc = C1e[tid];
        for (int m = 0; m < MT; m++) {
            float acc = cc;
#pragma unroll
            for (int p = 0; p < 8; p++) acc += xt[m * 18 + p] * wcol[p];
            h1[m * H1S + tid] = f2bf(fmaxf(acc, 0.f));
        }
    }
    __syncthreads();
    { // layer 2 WMMA with async-staged B strips, m-pair blocking
        const int mtA = wv & 3, nh = wv >> 2;
        const unsigned bOff = ldsoff(bufB);
        v16bf aA[8], aB[8];
#pragma unroll
        for (int kk = 0; kk < 8; kk++) {
            aA[kk] = load_frag_A(h1 + mtA * 16 * H1S,      H1S, lane, kk * 32);
            aB[kk] = load_frag_A(h1 + (mtA + 4) * 16 * H1S, H1S, lane, kk * 32);
        }
        // stage strips {0,1}
        for (int c = tid; c < 1024; c += 256)
            async_ld_b128(bOff + c * 16, w2T, c * 16);
        wait_async(); __syncthreads();
        for (int j = 0; j < 8; ++j) {
            if (j + 1 < 8)
                for (int c = tid; c < 1024; c += 256)
                    async_ld_b128(bOff + (unsigned)(((j + 1) & 1) * 16384 + c * 16),
                                  w2T, (unsigned)((j + 1) * 16384 + c * 16));
            const __bf16* bs = (const __bf16*)(bufB + (j & 1) * 16384 + nh * 8192);
            v8f acc0 = {}, acc1 = {};
#pragma unroll
            for (int kk = 0; kk < 8; kk++) {
                v16bf b = load_frag_Bs(bs, lane, kk * 32);
                acc0 = bf16_wmma(aA[kk], b, acc0);
                acc1 = bf16_wmma(aB[kk], b, acc1);
            }
            int c = (2 * j + nh) * 16 + (lane & 15);
            float s1 = 0.f, s2 = 0.f;
#pragma unroll
            for (int v = 0; v < 8; v++) {
                float q0 = acc0[v], q1 = acc1[v];
                s1 += q0 + q1; s2 += q0 * q0 + q1 * q1;
            }
            atomicAdd(&cs[c], s1);
            atomicAdd(&cq[c], s2);
            wait_async(); __syncthreads();
        }
    }
    atomicAdd(&gsum[tid], cs[tid]);
    atomicAdd(&gsq[tid],  cq[tid]);
}

// ---------------- kernel 5: finalize BN2 affine ----------------
__global__ __launch_bounds__(256) void pwq_fin2(
    const float* __restrict__ zs, const float* __restrict__ zq,
    const float* __restrict__ g2, const float* __restrict__ b2,
    float* __restrict__ a2, float* __restrict__ c2, float Binv)
{
    int j = threadIdx.x;
    float m = zs[j] * Binv;
    float var = zq[j] * Binv - m * m;
    float a = g2[j] * rsqrtf(var + BN_EPS);
    a2[j] = a; c2[j] = b2[j] - a * m;
}

// ---------------- kernel 6: fused MLP + spline ----------------
// dyn LDS layout (279808 B total):
//   xt   @0      : 128*18 f32          =   9216
//   h2   @9216   : 128*264 bf16        =  67584
//   jf   @76800  : 64*8 f32            =   2048
//   bufB @78848  : 64 KB strip buffer  =  65536
//   U    @144384 : union{ h1 128*264 bf16 (67584) , z3 64*529 f32 (135424) }
__global__ __launch_bounds__(256) void pwq_main(
    const float* __restrict__ x, const float* __restrict__ W1e, const float* __restrict__ C1e,
    const float* __restrict__ a2, const float* __restrict__ c2,
    const __bf16* __restrict__ w2T, const __bf16* __restrict__ w3T,
    const float* __restrict__ b3p, float* __restrict__ out)
{
    extern __shared__ char smem[];
    float*  xt   = (float*)smem;
    __bf16* h2   = (__bf16*)(smem + 9216);
    float*  jf   = (float*)(smem + 76800);
    char*   bufB = smem + 78848;
    __bf16* h1   = (__bf16*)(smem + 144384);
    float*  z3   = (float*)(smem + 144384);

    const int tid = threadIdx.x, lane = tid & 31, wv = tid >> 5;
    const int s0 = blockIdx.x * MT;
    const unsigned bOff = ldsoff(bufB);

    for (int i = tid; i < MT * 17; i += 256) {
        int s = i / 17, c = i % 17;
        xt[s * 18 + c] = x[(size_t)(s0 + s) * 17 + c];
    }
    __syncthreads();

    { // layer 1 (BN0+GEMM+BN1+ReLU folded): K=8, VALU
        float wcol[8];
#pragma unroll
        for (int p = 0; p < 8; p++) wcol[p] = W1e[p * HIDN + tid];
        float cc = C1e[tid];
        for (int m = 0; m < MT; m++) {
            float acc = cc;
#pragma unroll
            for (int p = 0; p < 8; p++) acc += xt[m * 18 + p] * wcol[p];
            h1[m * H1S + tid] = f2bf(fmaxf(acc, 0.f));
        }
    }
    __syncthreads();

    { // layer 2: 128x256x256 bf16 WMMA, async-staged B, m-pair blocking
        const int mtA = wv & 3, nh = wv >> 2;
        v16bf aA[8], aB[8];
#pragma unroll
        for (int kk = 0; kk < 8; kk++) {
            aA[kk] = load_frag_A(h1 + mtA * 16 * H1S,       H1S, lane, kk * 32);
            aB[kk] = load_frag_A(h1 + (mtA + 4) * 16 * H1S, H1S, lane, kk * 32);
        }
        for (int c = tid; c < 1024; c += 256)
            async_ld_b128(bOff + c * 16, w2T, c * 16);
        wait_async(); __syncthreads();
        for (int j = 0; j < 8; ++j) {
            if (j + 1 < 8)
                for (int c = tid; c < 1024; c += 256)
                    async_ld_b128(bOff + (unsigned)(((j + 1) & 1) * 16384 + c * 16),
                                  w2T, (unsigned)((j + 1) * 16384 + c * 16));
            const __bf16* bs = (const __bf16*)(bufB + (j & 1) * 16384 + nh * 8192);
            v8f acc0 = {}, acc1 = {};
#pragma unroll
            for (int kk = 0; kk < 8; kk++) {
                v16bf b = load_frag_Bs(bs, lane, kk * 32);
                acc0 = bf16_wmma(aA[kk], b, acc0);
                acc1 = bf16_wmma(aB[kk], b, acc1);
            }
            int c = (2 * j + nh) * 16 + (lane & 15);
            float av = a2[c], cv = c2[c];
            int rb0 = mtA * 16 + ((lane >> 4) << 3);
#pragma unroll
            for (int v = 0; v < 8; v++) {
                h2[(rb0 + v) * H1S + c]      = f2bf(fmaxf(av * acc0[v] + cv, 0.f));
                h2[(rb0 + 64 + v) * H1S + c] = f2bf(fmaxf(av * acc1[v] + cv, 0.f));
            }
            wait_async(); __syncthreads();
        }
    }
    __syncthreads();   // h1 dead; z3 may now reuse that region

    // layer 3 + spline, two halves of 64 samples (z3 aliases h1)
    for (int hh = 0; hh < 2; ++hh) {
        const int pr = wv & 1, nh3 = wv >> 1;     // m-pair, 4-way n split
        const int mt0 = 4 * hh + pr, mt1 = mt0 + 2;
        v16bf aA[8], aB[8];
#pragma unroll
        for (int kk = 0; kk < 8; kk++) {
            aA[kk] = load_frag_A(h2 + mt0 * 16 * H1S, H1S, lane, kk * 32);
            aB[kk] = load_frag_A(h2 + mt1 * 16 * H1S, H1S, lane, kk * 32);
        }
        for (int c = tid; c < 2048; c += 256)
            async_ld_b128(bOff + c * 16, w3T, c * 16);
        wait_async(); __syncthreads();
        for (int jj = 0; jj < 9; ++jj) {
            if (jj + 1 < 9)
                for (int c = tid; c < 2048; c += 256)
                    async_ld_b128(bOff + (unsigned)(((jj + 1) & 1) * 32768 + c * 16),
                                  w3T, (unsigned)((jj + 1) * 32768 + c * 16));
            const __bf16* bs = (const __bf16*)(bufB + (jj & 1) * 32768 + nh3 * 8192);
            v8f acc0 = {}, acc1 = {};
#pragma unroll
            for (int kk = 0; kk < 8; kk++) {
                v16bf b = load_frag_Bs(bs, lane, kk * 32);
                acc0 = bf16_wmma(aA[kk], b, acc0);
                acc1 = bf16_wmma(aB[kk], b, acc1);
            }
            int c = (4 * jj + nh3) * 16 + (lane & 15);
            if (c < OUTC) {
                float bv = b3p[c];
                int r0 = pr * 16 + ((lane >> 4) << 3);
#pragma unroll
                for (int v = 0; v < 8; v++) {
                    z3[(r0 + v) * Z3S + c]      = acc0[v] + bv;
                    z3[(r0 + 32 + v) * Z3S + c] = acc1[v] + bv;
                }
            }
            wait_async(); __syncthreads();
        }
        // spline on 64 samples of this half
        for (int pp = tid; pp < 64 * 8; pp += 256) {
            int sl = pp >> 3, t = pp & 7;
            int s  = 64 * hh + sl;
            const float* z = z3 + sl * Z3S + t * 65;
            float xb = fminf(xt[s * 18 + 8 + t], 1.0f - 1e-6f);
            float wn = 0.f, vn = 0.f;
            float vprev = __expf(z[0]);
            for (int i = 0; i < 32; i++) {
                float w_  = __expf(z[33 + i]);
                float vnx = __expf(z[i + 1]);
                wn += w_;
                vn += (vprev + vnx) * 0.5f * w_;
                vprev = vnx;
            }
            float inv_wn = 1.f / wn;
            float inv_vt = wn / vn;
            float cum = 0.f, vw = 0.f;
            float Wj = 1.f, Vj = 0.f, Vj1 = 0.f, shift = 0.f, W2 = 0.f;
            int found = 0;
            vprev = __expf(z[0]);
            for (int i = 0; i < 32; i++) {
                float w_  = __expf(z[33 + i]) * inv_wn;
                float vnx = __expf(z[i + 1]);
                float vi  = vprev * inv_vt, vi1 = vnx * inv_vt;
                float ncum = cum + w_;
                if (!found && (ncum > xb || i == 31)) {
                    found = 1; Wj = w_; Vj = vi; Vj1 = vi1; shift = vw; W2 = cum;
                }
                vw += (vi + vi1) * 0.5f * w_;
                cum = ncum; vprev = vnx;
            }
            float alpha = (xb - W2) / Wj;
            float yB = (0.5f * alpha * alpha * (Vj1 - Vj) + alpha * Vj) * Wj + shift;
            out[(size_t)(s0 + s) * 17 + 8 + t] = yB;
            jf[sl * 8 + t] = Vj + alpha * (Vj1 - Vj);
        }
        __syncthreads();
        if (tid < 64) {
            int sl = tid, s = 64 * hh + sl;
            float prod = 1.f;
#pragma unroll
            for (int t = 0; t < 8; t++) prod *= jf[sl * 8 + t];
            size_t o = (size_t)(s0 + s) * 17;
#pragma unroll
            for (int c = 0; c < 8; c++) out[o + c] = xt[s * 18 + c];  // xA passthrough
            out[o + 16] = xt[s * 18 + 16] * prod;                     // jac
        }
        __syncthreads();
    }
}

// ---------------- host launcher ----------------
extern "C" void kernel_launch(void* const* d_in, const int* in_sizes, int n_in,
                              void* d_out, int out_size, void* d_ws, size_t ws_size,
                              hipStream_t stream) {
    const float* x  = (const float*)d_in[0];
    const float* g0 = (const float*)d_in[1];
    const float* b0 = (const float*)d_in[2];
    const float* w1 = (const float*)d_in[3];
    const float* g1 = (const float*)d_in[4];
    const float* b1 = (const float*)d_in[5];
    const float* w2 = (const float*)d_in[6];
    const float* g2 = (const float*)d_in[7];
    const float* b2 = (const float*)d_in[8];
    const float* w3 = (const float*)d_in[9];
    const float* b3 = (const float*)d_in[10];
    float* out = (float*)d_out;
    const int B = in_sizes[0] / 17;

    char* wsb = (char*)d_ws;
    float* S0  = (float*)(wsb + 0);        // 8
    float* X0  = (float*)(wsb + 256);      // 64
    float* W1e = (float*)(wsb + 1024);     // 8*256
    float* C1e = (float*)(wsb + 9216);     // 256
    float* ZS  = (float*)(wsb + 10240);    // 256
    float* ZQ  = (float*)(wsb + 11264);    // 256
    float* A2  = (float*)(wsb + 12288);    // 256
    float* C2  = (float*)(wsb + 13312);    // 256
    __bf16* w2T = (__bf16*)(wsb + 16384);                   // 256*256 bf16 (128 KB)
    __bf16* w3T = (__bf16*)(wsb + 16384 + 131072);          // 576*256 bf16 (288 KB)
    float*  B3P = (float*)(wsb + 16384 + 131072 + 294912);  // 576 f32

    hipMemsetAsync(wsb, 0, 14336, stream);  // zero stat accumulators each call

    pwq_prepw <<<256, 256, 0, stream>>>(w2, w3, b3, w2T, w3T, B3P);
    pwq_stats0<<<64,  256, 0, stream>>>(x, B, S0, X0);
    pwq_prep01<<<1,   256, 0, stream>>>(S0, X0, g0, b0, w1, g1, b1, W1e, C1e, 1.f / (float)B);

    const size_t smem2 = 9216 + 67584 + 32768 + 2048;        // 111616
    (void)hipFuncSetAttribute((const void*)pwq_stats2,
                              hipFuncAttributeMaxDynamicSharedMemorySize, (int)smem2);
    pwq_stats2<<<B / MT, 256, smem2, stream>>>(x, W1e, C1e, w2T, ZS, ZQ);
    pwq_fin2  <<<1, 256, 0, stream>>>(ZS, ZQ, g2, b2, A2, C2, 1.f / (float)B);

    const size_t smem = 144384 + 135424;                     // 279808
    (void)hipFuncSetAttribute((const void*)pwq_main,
                              hipFuncAttributeMaxDynamicSharedMemorySize, (int)smem);
    pwq_main<<<B / MT, 256, smem, stream>>>(x, W1e, C1e, A2, C2, w2T, w3T, B3P, out);
}
